// ProteinFeatureEncoder_84138409328672
// MI455X (gfx1250) — compile-verified, matrix operand order checked by reading, not run on hardware
//
#include <hip/hip_runtime.h>
#include <hip/hip_bf16.h>
#include <math.h>

typedef __attribute__((ext_vector_type(16))) _Float16 v16h;
typedef __attribute__((ext_vector_type(8)))  float    v8f;

#define NTOT 768              // N_RES*N_ATOMS per batch = 768
#define NB 2
#define NODE_COUNT (NB*NTOT)                     // 1536
#define EDGE_ELEMS (2ull*768ull*768ull*64ull)    // 75497472
#define NODE_ELEMS (2*768*128)                   // 196608

__device__ __forceinline__ float fast_silu(float x) {
    // x * sigmoid(x) with hardware v_rcp_f32 (avoids IEEE div expansion)
    return x * __builtin_amdgcn_rcpf(1.0f + __expf(-x));
}

// ---------------------------------------------------------------------------
// Kernel 1: mutation MLP  (384 residues, 124 -> 64 silu -> 32, * mask)
// ---------------------------------------------------------------------------
__global__ void mut_kernel(const float* __restrict__ aa_emb,
                           const float* __restrict__ aa_props,
                           const int*   __restrict__ wt_idx,
                           const int*   __restrict__ mut_idx,
                           const float* __restrict__ W1, const float* __restrict__ b1,
                           const float* __restrict__ W2, const float* __restrict__ b2,
                           const float* __restrict__ mut_mask,
                           float* __restrict__ mut_feat) {
    __shared__ float in[124];
    __shared__ float h[64];
    int r = blockIdx.x;           // 0..383
    int t = threadIdx.x;          // 0..63
    int wt = wt_idx[r], mu = mut_idx[r];
    if (t < 32)             in[t]            = aa_emb[wt*32 + t];
    else                    in[t]            = aa_emb[mu*32 + (t-32)];
    if (t < 30)             in[64 + t]       = aa_props[wt*30 + t];
    if (t >= 32 && t < 62)  in[94 + (t-32)]  = aa_props[mu*30 + (t-32)];
    __syncthreads();
    float acc = b1[t];
    #pragma unroll 4
    for (int k = 0; k < 124; ++k) acc += in[k] * W1[k*64 + t];
    h[t] = fast_silu(acc);
    __syncthreads();
    if (t < 32) {
        float o = b2[t];
        #pragma unroll 4
        for (int k = 0; k < 64; ++k) o += h[k] * W2[k*32 + t];
        mut_feat[r*32 + t] = o * mut_mask[r];
    }
}

// ---------------------------------------------------------------------------
// Kernel 2: node MLP (1536 atom rows, 126 -> 128 silu -> 128, * atom_mask)
//           also writes coords_flat and atom_mask_flat outputs.
// node_in = [wt_e 32 | wt_p 30 | mut_feat 32 | atom_emb 16 | pos_emb 16]
// ---------------------------------------------------------------------------
__global__ void node_kernel(const float* __restrict__ aa_emb,
                            const float* __restrict__ aa_props,
                            const float* __restrict__ atom_emb,
                            const float* __restrict__ pos_emb,
                            const int*   __restrict__ wt_idx,
                            const int*   __restrict__ res_idx,
                            const float* __restrict__ mut_feat,
                            const float* __restrict__ W1, const float* __restrict__ b1,
                            const float* __restrict__ W2, const float* __restrict__ b2,
                            const float* __restrict__ atom_mask,
                            const float* __restrict__ coords,
                            float* __restrict__ node_out,
                            float* __restrict__ coords_out,
                            float* __restrict__ mask_out) {
    __shared__ float in[128];
    __shared__ float h[128];
    int a  = blockIdx.x;          // flat atom row: b*768 + res*4 + atom
    int r  = a >> 2;              // flat residue: b*192 + res
    int at = a & 3;               // atom type 0..3
    int t  = threadIdx.x;         // 0..127
    int wt = wt_idx[r];
    if      (t < 32)  in[t] = aa_emb[wt*32 + t];
    else if (t < 62)  in[t] = aa_props[wt*30 + (t-32)];
    else if (t < 94)  in[t] = mut_feat[r*32 + (t-62)];
    else if (t < 110) in[t] = atom_emb[at*16 + (t-94)];
    else if (t < 126) in[t] = pos_emb[res_idx[a]*16 + (t-110)];
    else              in[t] = 0.0f;
    __syncthreads();
    float acc = b1[t];
    #pragma unroll 4
    for (int k = 0; k < 126; ++k) acc += in[k] * W1[k*128 + t];
    h[t] = fast_silu(acc);
    __syncthreads();
    float o = b2[t];
    #pragma unroll 4
    for (int k = 0; k < 128; ++k) o += h[k] * W2[k*128 + t];
    float msk = atom_mask[a];
    node_out[a*128 + t] = o * msk;
    if (t < 3)  coords_out[a*3 + t] = coords[a*3 + t];
    if (t == 3) mask_out[a] = msk;
}

// ---------------------------------------------------------------------------
// Kernel 3: edge MLP (dominant).  Per wave: 16 pairs -> RBF(32) ->
//   GEMM1 (4 independent v_wmma_f32_16x16x32_f16 back-to-back) -> silu ->
//   LDS transpose (C/D layout -> A layout) ->
//   GEMM2 (K=64: 4 independent + 4 accumulation WMMAs) -> bias+adj -> store
// Weight B-tiles register-resident; independent accumulators so WMMA->VALU
// hazard NOPs are hidden behind other WMMAs.
// ---------------------------------------------------------------------------
__global__ void __launch_bounds__(256, 1)
edge_kernel(const float* __restrict__ coords,      // [1536][3]
            const float* __restrict__ atom_mask,   // [1536]
            const float* __restrict__ W1,          // [32][64]
            const float* __restrict__ b1,          // [64]
            const float* __restrict__ W2,          // [64][64]
            const float* __restrict__ b2,          // [64]
            float* __restrict__ edge_out) {
    __shared__ _Float16 h_lds[8][16*64];    // per-wave hidden tile (2KB each)
    __shared__ float    adj_lds[8][16];     // per-wave adjacency per row

    const int lane = threadIdx.x & 31;
    const int wib  = threadIdx.x >> 5;                       // wave in block
    const int waveId   = blockIdx.x * (blockDim.x >> 5) + wib;
    const int numWaves = gridDim.x * (blockDim.x >> 5);
    const int half  = lane >> 4;       // 0: K 0-7/16-23, 1: K 8-15/24-31
    const int l16   = lane & 15;
    const int kbase = half * 8;

    // ---- preload W1 B-tiles (32x16 each, 4 tiles covering 64 hidden cols) ----
    v16h B1[4];
    float bias1[4], bias2[4];
    #pragma unroll
    for (int t = 0; t < 4; ++t) {
        const int n = t*16 + l16;
        v16h bb;
        #pragma unroll
        for (int e = 0; e < 8; ++e) {
            bb[e]     = (_Float16)W1[(kbase + e)      * 64 + n];
            bb[e + 8] = (_Float16)W1[(kbase + 16 + e) * 64 + n];
        }
        B1[t] = bb;
        bias1[t] = b1[n];
        bias2[t] = b2[n];
    }
    // ---- preload W2 B-tiles: 2 K-chunks of 32 x 4 N-tiles ----
    v16h B2[2][4];
    #pragma unroll
    for (int c = 0; c < 2; ++c) {
        #pragma unroll
        for (int t = 0; t < 4; ++t) {
            const int n = t*16 + l16;
            v16h bb;
            #pragma unroll
            for (int e = 0; e < 8; ++e) {
                bb[e]     = (_Float16)W2[(c*32 + kbase + e)      * 64 + n];
                bb[e + 8] = (_Float16)W2[(c*32 + kbase + 16 + e) * 64 + n];
            }
            B2[c][t] = bb;
        }
    }

    const int numTiles = (NB * NTOT * NTOT) / 16;   // 73728
    const float inv2g2 = 1.28f;          // 1/(2*gamma^2), gamma = 20/32
    const float cstep  = 20.0f / 31.0f;  // linspace(0,20,32) step
    _Float16* hl = h_lds[wib];

    for (int tile = waveId; tile < numTiles; tile += numWaves) {
        // This lane's A-matrix row = pair index prow
        const int prow = tile*16 + l16;
        const int b    = prow / (NTOT*NTOT);
        const int rem  = prow - b*(NTOT*NTOT);
        const int i    = rem / NTOT;
        const int j    = rem - i*NTOT;
        const int ri   = b*NTOT + i, rj = b*NTOT + j;

        const float dx = coords[ri*3+0] - coords[rj*3+0];
        const float dy = coords[ri*3+1] - coords[rj*3+1];
        const float dz = coords[ri*3+2] - coords[rj*3+2];
        const float dist = sqrtf(dx*dx + dy*dy + dz*dz + 1e-8f);

        float adj = (dist <= 7.5f) ? 1.0f : 0.0f;
        adj *= atom_mask[ri] * atom_mask[rj];
        if (i == j) adj = 0.0f;
        if (half == 0) adj_lds[wib][l16] = adj;

        // ---- build 16x32 f16 RBF A-matrix (ISA A-layout) ----
        v16h A;
        #pragma unroll
        for (int e = 0; e < 8; ++e) {
            const float d0 = dist - (float)(kbase + e)      * cstep;
            const float d1 = dist - (float)(kbase + 16 + e) * cstep;
            A[e]     = (_Float16)__expf(-d0*d0 * inv2g2);
            A[e + 8] = (_Float16)__expf(-d1*d1 * inv2g2);
        }

        // ---- GEMM1: 4 independent WMMAs back-to-back (hazards overlap) ----
        v8f H[4];
        #pragma unroll
        for (int t = 0; t < 4; ++t) {
            v8f z = {};
            H[t] = __builtin_amdgcn_wmma_f32_16x16x32_f16(
                false, A, false, B1[t], (short)0, z, false, false);
        }
        // ---- bias + silu, stash hidden into LDS in [row][col] f16 ----
        #pragma unroll
        for (int t = 0; t < 4; ++t) {
            const float bia = bias1[t];
            #pragma unroll
            for (int v = 0; v < 8; ++v) {
                const float s = fast_silu(H[t][v] + bia);
                hl[(v + half*8)*64 + t*16 + l16] = (_Float16)s;
            }
        }
        __builtin_amdgcn_wave_barrier();   // same-wave, in-order LDS: sched fence

        // ---- adjacency per output row, 8 contiguous LDS floats ----
        float adjv[8];
        #pragma unroll
        for (int v = 0; v < 8; ++v) adjv[v] = adj_lds[wib][half*8 + v];

        // ---- re-load hidden in A-layout for GEMM2 (K=64 -> two 32-chunks) ----
        v16h A2[2];
        #pragma unroll
        for (int c = 0; c < 2; ++c) {
            v16h aa;
            #pragma unroll
            for (int e = 0; e < 8; ++e) {
                aa[e]     = hl[l16*64 + c*32 + kbase + e];
                aa[e + 8] = hl[l16*64 + c*32 + kbase + 16 + e];
            }
            A2[c] = aa;
        }

        // ---- GEMM2: 4 independent, then 4 accumulation WMMAs ----
        v8f ACC[4];
        #pragma unroll
        for (int t = 0; t < 4; ++t) {
            v8f z = {};
            ACC[t] = __builtin_amdgcn_wmma_f32_16x16x32_f16(
                false, A2[0], false, B2[0][t], (short)0, z, false, false);
        }
        #pragma unroll
        for (int t = 0; t < 4; ++t) {
            ACC[t] = __builtin_amdgcn_wmma_f32_16x16x32_f16(
                false, A2[1], false, B2[1][t], (short)0, ACC[t], false, false);
        }
        // ---- bias + adjacency mask + coalesced store ----
        #pragma unroll
        for (int t = 0; t < 4; ++t) {
            const float bia = bias2[t];
            #pragma unroll
            for (int v = 0; v < 8; ++v) {
                const float val = (ACC[t][v] + bia) * adjv[v];
                const size_t p = (size_t)tile*16 + (v + half*8);
                edge_out[p*64 + t*16 + l16] = val;
            }
        }
        __builtin_amdgcn_wave_barrier();   // keep next iter's LDS writes after reads
    }
}

// ---------------------------------------------------------------------------
extern "C" void kernel_launch(void* const* d_in, const int* in_sizes, int n_in,
                              void* d_out, int out_size, void* d_ws, size_t ws_size,
                              hipStream_t stream) {
    const float* coords    = (const float*)d_in[0];
    const float* mut_mask  = (const float*)d_in[1];
    const float* atom_mask = (const float*)d_in[2];
    const float* aa_emb    = (const float*)d_in[3];
    const float* atom_emb  = (const float*)d_in[4];
    const float* pos_emb   = (const float*)d_in[5];
    const float* aa_props  = (const float*)d_in[6];
    const float* mut_W1    = (const float*)d_in[7];
    const float* mut_b1    = (const float*)d_in[8];
    const float* mut_W2    = (const float*)d_in[9];
    const float* mut_b2    = (const float*)d_in[10];
    const float* node_W1   = (const float*)d_in[11];
    const float* node_b1   = (const float*)d_in[12];
    const float* node_W2   = (const float*)d_in[13];
    const float* node_b2   = (const float*)d_in[14];
    const float* edge_W1   = (const float*)d_in[15];
    const float* edge_b1   = (const float*)d_in[16];
    const float* edge_W2   = (const float*)d_in[17];
    const float* edge_b2   = (const float*)d_in[18];
    const int*   wt_idx    = (const int*)d_in[19];
    const int*   mut_idx   = (const int*)d_in[20];
    const int*   res_idx   = (const int*)d_in[21];

    float* out        = (float*)d_out;
    float* node_out   = out;                        // [2,768,128]
    float* edge_out   = out + NODE_ELEMS;           // [2,768,768,64]
    float* coords_out = edge_out + EDGE_ELEMS;      // [2,768,3]
    float* mask_out   = coords_out + NB*NTOT*3;     // [2,768]
    float* mut_feat   = (float*)d_ws;               // [384,32] scratch

    mut_kernel<<<384, 64, 0, stream>>>(aa_emb, aa_props, wt_idx, mut_idx,
                                       mut_W1, mut_b1, mut_W2, mut_b2,
                                       mut_mask, mut_feat);
    node_kernel<<<NODE_COUNT, 128, 0, stream>>>(aa_emb, aa_props, atom_emb, pos_emb,
                                                wt_idx, res_idx, mut_feat,
                                                node_W1, node_b1, node_W2, node_b2,
                                                atom_mask, coords,
                                                node_out, coords_out, mask_out);
    // 73728 tiles of 16 pairs; 1152 blocks x 8 waves -> 8 tiles per wave
    edge_kernel<<<1152, 256, 0, stream>>>(coords, atom_mask,
                                          edge_W1, edge_b1, edge_W2, edge_b2,
                                          edge_out);
}
